// GAT_75273596830284
// MI455X (gfx1250) — matrix operand order repaired and checked by reading, not compiled, not run on hardware
//
#include <hip/hip_runtime.h>
#include <math.h>

typedef __bf16 bf16_t;
typedef __attribute__((ext_vector_type(16))) __bf16 v16bf;
typedef __attribute__((ext_vector_type(8)))  float  v8f;

#define NFEAT   256
#define HID     32
#define NHEAD   8
#define NCLASS  40
#define L2PAD   48         // pad layer-2 cols 40 -> 48 (3 x 16 tiles)
#define NEG_SLOPE 0.2f

// ---------------------------------------------------------------- utilities

__device__ __forceinline__ void atomicMaxF(float* addr, float val) {
  // sign-split trick: works with init bits 0xFFFFFFFF (NaN-pattern) since
  // int(0xFFFFFFFF) = -1 (< any positive-float bits) and uint 0xFFFFFFFF is max.
  if (val >= 0.f) atomicMax((int*)addr, __float_as_int(val));
  else            atomicMin((unsigned int*)addr, __float_as_uint(val));
}

__device__ __forceinline__ void edge_nodes(const int* __restrict__ ei, int E,
                                           int e, int& s, int& d) {
  if (e < E) { s = ei[e]; d = ei[E + e]; }   // edge_index rows [2,E]
  else       { s = e - E; d = s; }           // appended self loop
}

// ---------------------------------------------------------------- conversions

__global__ void f32_to_bf16_k(const float* __restrict__ in,
                              bf16_t* __restrict__ out, int n) {
  int i = blockIdx.x * blockDim.x + threadIdx.x;
  if (i < n) out[i] = (bf16_t)in[i];
}

// W[k, Nc] (row-major) -> Wt[n, K] bf16, zero-padded to NcPad columns-of-W rows
__global__ void convT_pad_k(const float* __restrict__ W, bf16_t* __restrict__ Wt,
                            int K, int Nc, int NcPad) {
  int i = blockIdx.x * blockDim.x + threadIdx.x;   // i = n*K + k
  if (i >= NcPad * K) return;
  int n = i / K, k = i - n * K;
  Wt[i] = (n < Nc) ? (bf16_t)W[(size_t)k * Nc + n] : (bf16_t)0.f;
}

// ---------------------------------------------------------------- WMMA GEMM
// C[M, ldc] = A[M, K](bf16) x Bt[Ncols, K](bf16, transposed weight), f32 accum.
// One wave32 per 16x16 tile. M must be a multiple of 16 (50000 = 3125*16).
__global__ void wmma_gemm_bf16_k(const bf16_t* __restrict__ A,
                                 const bf16_t* __restrict__ Bt,
                                 float* __restrict__ C,
                                 int Mtiles, int K, int NT, int ldc) {
  int wave = (blockIdx.x * blockDim.x + threadIdx.x) >> 5;
  int lane = threadIdx.x & 31;
  if (wave >= Mtiles * NT) return;             // wave-uniform: EXEC stays all-1s
  int tm = wave / NT, tn = wave - tm * NT;
  int half = lane >> 4, l16 = lane & 15;

  const bf16_t* arow = A  + (size_t)(tm * 16 + l16) * K;  // A row m = l16
  const bf16_t* brow = Bt + (size_t)(tn * 16 + l16) * K;  // B col n = l16

  v8f acc = {};
#pragma unroll
  for (int kb = 0; kb < 256; kb += 32) {
    v16bf a, b;
    // A 16x32 bf16 fragment: element j holds K = (j/8)*16 + half*8 + j%8
#pragma unroll
    for (int j = 0; j < 8; ++j) {
      a[j]     = arow[kb + half * 8 + j];
      a[j + 8] = arow[kb + 16 + half * 8 + j];
    }
    // B 32x16 bf16 fragment: element j holds K = (j/8)*16 + half*8 + j%8
#pragma unroll
    for (int j = 0; j < 16; ++j) b[j] = brow[kb + (j >> 3) * 16 + half * 8 + (j & 7)];

    acc = __builtin_amdgcn_wmma_f32_16x16x32_bf16(
        /*neg_a=*/false, a, /*neg_b=*/false, b,
        /*c_mod=*/(short)0, acc, /*reuse_a=*/false, /*reuse_b=*/false);
  }
  // C fragment: VGPR r -> M = half*8 + r, N = l16
  float* crow = C + (size_t)(tm * 16 + half * 8) * ldc + tn * 16 + l16;
#pragma unroll
  for (int r = 0; r < 8; ++r) crow[(size_t)r * ldc] = acc[r];
}

// ---------------------------------------------------------------- attention dots
// asrc[n,h] = sum_c Hm[n, h*C + c] * a_src[h, c]   (ditto adst)
__global__ void att_dots_k(const float* __restrict__ Hm, int ld, int C, int NH,
                           const float* __restrict__ a_src,
                           const float* __restrict__ a_dst,
                           float* __restrict__ asrc, float* __restrict__ adst,
                           int N) {
  int i = blockIdx.x * blockDim.x + threadIdx.x;   // i = n*NH + h
  if (i >= N * NH) return;
  int n = i / NH, h = i - n * NH;
  const float* hp = Hm + (size_t)n * ld + h * C;
  float s0 = 0.f, s1 = 0.f;
  for (int c = 0; c < C; ++c) {
    float v = hp[c];
    s0 += v * a_src[h * C + c];
    s1 += v * a_dst[h * C + c];
  }
  asrc[i] = s0;
  adst[i] = s1;
}

// ---------------------------------------------------------------- edge passes

__global__ void edge_max_k(const int* __restrict__ ei, int E, int N, int NH,
                           const float* __restrict__ asrc,
                           const float* __restrict__ adst,
                           float* __restrict__ m) {
  int i = blockIdx.x * blockDim.x + threadIdx.x;
  if (i >= (E + N) * NH) return;
  int e = i / NH, h = i - e * NH;
  int s, d; edge_nodes(ei, E, e, s, d);
  float v = asrc[s * NH + h] + adst[d * NH + h];
  v = v > 0.f ? v : NEG_SLOPE * v;
  atomicMaxF(&m[d * NH + h], v);
}

__global__ void edge_sum_k(const int* __restrict__ ei, int E, int N, int NH,
                           const float* __restrict__ asrc,
                           const float* __restrict__ adst,
                           const float* __restrict__ m,
                           float* __restrict__ ssum) {
  int i = blockIdx.x * blockDim.x + threadIdx.x;
  if (i >= (E + N) * NH) return;
  int e = i / NH, h = i - e * NH;
  int s, d; edge_nodes(ei, E, e, s, d);
  float v = asrc[s * NH + h] + adst[d * NH + h];
  v = v > 0.f ? v : NEG_SLOPE * v;
  atomicAdd(&ssum[d * NH + h], __expf(v - m[d * NH + h]));
}

// layer 1 weighted scatter: one block (256 threads = full feature row) per edge
__global__ void scatter1_k(const int* __restrict__ ei, int E, int N,
                           const float* __restrict__ asrc,
                           const float* __restrict__ adst,
                           const float* __restrict__ m,
                           const float* __restrict__ ssum,
                           const float* __restrict__ H1,
                           float* __restrict__ O1) {
  int e = blockIdx.x;
  int s, d; edge_nodes(ei, E, e, s, d);
  int c = threadIdx.x;            // 0..255
  int h = c >> 5;                 // head = c / 32
  float v = asrc[s * NHEAD + h] + adst[d * NHEAD + h];
  v = v > 0.f ? v : NEG_SLOPE * v;
  float a = __expf(v - m[d * NHEAD + h]) / (ssum[d * NHEAD + h] + 1e-16f);
  atomicAdd(&O1[(size_t)d * 256 + c], H1[(size_t)s * 256 + c] * a);
}

// layer 2 weighted scatter: thread per (edge, class)
__global__ void scatter2_k(const int* __restrict__ ei, int E, int N,
                           const float* __restrict__ asrc,
                           const float* __restrict__ adst,
                           const float* __restrict__ m,
                           const float* __restrict__ ssum,
                           const float* __restrict__ H2,
                           float* __restrict__ O2, int total) {
  int i = blockIdx.x * blockDim.x + threadIdx.x;
  if (i >= total) return;
  int e = i / NCLASS, c = i - e * NCLASS;
  int s, d; edge_nodes(ei, E, e, s, d);
  float v = asrc[s] + adst[d];
  v = v > 0.f ? v : NEG_SLOPE * v;
  float a = __expf(v - m[d]) / (ssum[d] + 1e-16f);
  atomicAdd(&O2[(size_t)d * NCLASS + c], H2[(size_t)s * L2PAD + c] * a);
}

// ---------------------------------------------------------------- activations

__global__ void bias_elu_bf16_k(const float* __restrict__ O1,
                                const float* __restrict__ b1,
                                bf16_t* __restrict__ Hb, int total) {
  int i = blockIdx.x * blockDim.x + threadIdx.x;
  if (i >= total) return;
  float v = O1[i] + b1[i & 255];
  v = v > 0.f ? v : (__expf(v) - 1.f);    // ELU(alpha=1)
  Hb[i] = (bf16_t)v;
}

__global__ void finalize_k(const float* __restrict__ O2,
                           const float* __restrict__ b2,
                           float* __restrict__ out, int N) {
  int n = blockIdx.x * blockDim.x + threadIdx.x;
  if (n >= N) return;
  float v[NCLASS];
  float mx = -3.0e38f;
  for (int c = 0; c < NCLASS; ++c) {
    v[c] = O2[(size_t)n * NCLASS + c] + b2[c];
    mx = fmaxf(mx, v[c]);
  }
  float ssum = 0.f;
  for (int c = 0; c < NCLASS; ++c) ssum += __expf(v[c] - mx);
  float lse = mx + __logf(ssum);
  float* o_final = out + (size_t)n * NCLASS;
  float* o_lsm   = out + (size_t)N * NCLASS + (size_t)n * NCLASS;
  for (int c = 0; c < NCLASS; ++c) { o_final[c] = v[c]; o_lsm[c] = v[c] - lse; }
}

// ---------------------------------------------------------------- launch

extern "C" void kernel_launch(void* const* d_in, const int* in_sizes, int n_in,
                              void* d_out, int out_size, void* d_ws, size_t ws_size,
                              hipStream_t stream) {
  const float* x   = (const float*)d_in[0];
  const int*   ei  = (const int*)  d_in[1];
  // d_in[2] = edge_attr, unused by the model
  const float* W1  = (const float*)d_in[3];
  const float* as1 = (const float*)d_in[4];
  const float* ad1 = (const float*)d_in[5];
  const float* b1  = (const float*)d_in[6];
  const float* W2  = (const float*)d_in[7];
  const float* as2 = (const float*)d_in[8];
  const float* ad2 = (const float*)d_in[9];
  const float* b2  = (const float*)d_in[10];

  const int N  = in_sizes[0] / NFEAT;   // 50000
  const int E  = in_sizes[1] / 2;       // 800000
  const int Et = E + N;                 // edges + self loops
  const int Mt = N / 16;                // 3125 row tiles (exact)

  // workspace carve-out
  char* p = (char*)d_ws;
  auto alloc = [&](size_t bytes) -> char* {
    char* r = p; p += (bytes + 255) & ~(size_t)255; return r;
  };
  bf16_t* Xbf   = (bf16_t*)alloc((size_t)N * NFEAT * 2);
  bf16_t* Wt1   = (bf16_t*)alloc((size_t)256 * 256 * 2);
  bf16_t* Wt2   = (bf16_t*)alloc((size_t)L2PAD * 256 * 2);
  float*  H1    = (float*) alloc((size_t)N * 256 * 4);
  float*  O1    = (float*) alloc((size_t)N * 256 * 4);
  bf16_t* H1b   = (bf16_t*)alloc((size_t)N * 256 * 2);
  float*  H2    = (float*) alloc((size_t)N * L2PAD * 4);
  float*  O2    = (float*) alloc((size_t)N * NCLASS * 4);
  float*  asrc1 = (float*) alloc((size_t)N * NHEAD * 4);
  float*  adst1 = (float*) alloc((size_t)N * NHEAD * 4);
  float*  m1    = (float*) alloc((size_t)N * NHEAD * 4);
  float*  s1v   = (float*) alloc((size_t)N * NHEAD * 4);
  float*  asrc2 = (float*) alloc((size_t)N * 4);
  float*  adst2 = (float*) alloc((size_t)N * 4);
  float*  m2    = (float*) alloc((size_t)N * 4);
  float*  s2v   = (float*) alloc((size_t)N * 4);

  // init accumulators (graph-capturable async memsets)
  hipMemsetAsync(O1,  0,    (size_t)N * 256 * 4,    stream);
  hipMemsetAsync(O2,  0,    (size_t)N * NCLASS * 4, stream);
  hipMemsetAsync(s1v, 0,    (size_t)N * NHEAD * 4,  stream);
  hipMemsetAsync(s2v, 0,    (size_t)N * 4,          stream);
  hipMemsetAsync(m1,  0xFF, (size_t)N * NHEAD * 4,  stream);  // -"inf" bits
  hipMemsetAsync(m2,  0xFF, (size_t)N * 4,          stream);

  const int B = 256;
  // ---- precision / layout prep
  f32_to_bf16_k<<<(N * NFEAT + B - 1) / B, B, 0, stream>>>(x, Xbf, N * NFEAT);
  convT_pad_k  <<<(256 * 256 + B - 1) / B, B, 0, stream>>>(W1, Wt1, 256, 256, 256);
  convT_pad_k  <<<(L2PAD * 256 + B - 1) / B, B, 0, stream>>>(W2, Wt2, 256, NCLASS, L2PAD);

  // ---- layer 1: h = x @ W1  (bf16 WMMA, f32 accum)
  { int waves = Mt * 16; wmma_gemm_bf16_k<<<(waves + 7) / 8, 256, 0, stream>>>(
        Xbf, Wt1, H1, Mt, 256, 16, 256); }
  att_dots_k<<<(N * NHEAD + B - 1) / B, B, 0, stream>>>(
      H1, 256, HID, NHEAD, as1, ad1, asrc1, adst1, N);
  edge_max_k<<<(Et * NHEAD + B - 1) / B, B, 0, stream>>>(ei, E, N, NHEAD, asrc1, adst1, m1);
  edge_sum_k<<<(Et * NHEAD + B - 1) / B, B, 0, stream>>>(ei, E, N, NHEAD, asrc1, adst1, m1, s1v);
  scatter1_k<<<Et, 256, 0, stream>>>(ei, E, N, asrc1, adst1, m1, s1v, H1, O1);
  bias_elu_bf16_k<<<(N * 256 + B - 1) / B, B, 0, stream>>>(O1, b1, H1b, N * 256);

  // ---- layer 2: h2 = elu(out1) @ W2  (padded to 48 cols)
  { int waves = Mt * 3; wmma_gemm_bf16_k<<<(waves + 7) / 8, 256, 0, stream>>>(
        H1b, Wt2, H2, Mt, 256, 3, L2PAD); }
  att_dots_k<<<(N + B - 1) / B, B, 0, stream>>>(
      H2, L2PAD, NCLASS, 1, as2, ad2, asrc2, adst2, N);
  edge_max_k<<<(Et + B - 1) / B, B, 0, stream>>>(ei, E, N, 1, asrc2, adst2, m2);
  edge_sum_k<<<(Et + B - 1) / B, B, 0, stream>>>(ei, E, N, 1, asrc2, adst2, m2, s2v);
  { int tot2 = Et * NCLASS;
    scatter2_k<<<(tot2 + B - 1) / B, B, 0, stream>>>(
        ei, E, N, asrc2, adst2, m2, s2v, H2, O2, tot2); }

  // ---- bias + log_softmax, write both outputs
  finalize_k<<<(N + B - 1) / B, B, 0, stream>>>(O2, b2, (float*)d_out, N);
}
